// CausalSelfAttention_72971494359577
// MI455X (gfx1250) — matrix operand order repaired and checked by reading, not compiled, not run on hardware
//
#include <hip/hip_runtime.h>

typedef __attribute__((ext_vector_type(16))) _Float16 v16h;
typedef __attribute__((ext_vector_type(8)))  _Float16 v8h;
typedef __attribute__((ext_vector_type(4)))  _Float16 v4h;
typedef __attribute__((ext_vector_type(8)))  float    v8f;
typedef __attribute__((ext_vector_type(4)))  unsigned int u32x4;
typedef __attribute__((ext_vector_type(8)))  int i32x8;
typedef __attribute__((ext_vector_type(4)))  int i32x4;

#define DIMX   1024
#define NHEADS 16
#define HDIM   64
#define BBATCH 4
#define LSEQ   2048
#define BL     (BBATCH * LSEQ)      // 8192 rows
#define EQKV   (3 * DIMX)           // 3072

// ---------------------------------------------------------------------------
// WMMA helpers (CDNA5 wave32, V_WMMA_F32_16X16X32_F16)
// ---------------------------------------------------------------------------
__device__ __forceinline__ v8f wmma32(v16h a, v16h b, v8f c) {
  return __builtin_amdgcn_wmma_f32_16x16x32_f16(
      /*neg_a=*/false, a, /*neg_b=*/false, b,
      /*c_mod=*/(short)0, c, /*reuse_a=*/false, /*reuse_b=*/false);
}

// A fragment (16x32 f16): lane (m = lane&15, hf = lane>>4) holds
// K = [8*hf .. 8*hf+7] and [16+8*hf .. 16+8*hf+7] of row m (row contiguous).
__device__ __forceinline__ v16h load_a_frag(const _Float16* rowbase, int hf) {
  v8h lo = *(const v8h*)(rowbase + 8 * hf);
  v8h hi = *(const v8h*)(rowbase + 16 + 8 * hf);
  v16h a;
#pragma unroll
  for (int i = 0; i < 8; ++i) { a[i] = lo[i]; a[i + 8] = hi[i]; }
  return a;
}

// B fragment (32x16 f16): lane (n = lane&15, hf = lane>>4) holds
// K = [8*hf .. 8*hf+7] and [16+8*hf .. 16+8*hf+7] of column n (column contiguous in memory/LDS).
__device__ __forceinline__ v16h load_b_frag(const _Float16* colbase, int hf) {
  v8h lo = *(const v8h*)(colbase + 8 * hf);
  v8h hi = *(const v8h*)(colbase + 16 + 8 * hf);
  v16h b;
#pragma unroll
  for (int i = 0; i < 8; ++i) { b[i] = lo[i]; b[i + 8] = hi[i]; }
  return b;
}

// ---------------------------------------------------------------------------
// Tensor Data Mover: 2D tile (data_size = 8B) global -> LDS.
// Descriptor per CDNA5 ISA 8.3-8.6: group0 = {count/lds_addr/global_addr/type},
// group1 = {data_size, tensor dims, tile dims, dim0 stride}. Groups 2/3 zero
// (2D tensor). Issued by one wave; completion via TENSORcnt.
// ---------------------------------------------------------------------------
__device__ __forceinline__ void tdm_load_tile(
    void* lds_dst, const void* gsrc,
    unsigned tile_d0, unsigned tile_d1,
    unsigned tensor_d0, unsigned tensor_d1,
    unsigned long long d0_stride_units) {
  const unsigned lds_addr = (unsigned)(uintptr_t)lds_dst;       // LDS byte offset
  const unsigned long long ga = (unsigned long long)(uintptr_t)gsrc;
  u32x4 g0;
  g0[0] = 1u;                                                   // count=1, user desc
  g0[1] = lds_addr;                                             // bits 63:32
  g0[2] = (unsigned)(ga & 0xffffffffull);                       // global_addr[31:0]
  g0[3] = (unsigned)((ga >> 32) & 0x01ffffffull) | (2u << 30);  // addr[56:32] | type=2
  i32x8 g1;
  g1[0] = (int)(3u << 16);                                      // wg_mask=0, data_size=8B
  g1[1] = (int)((tensor_d0 & 0xffffu) << 16);                   // tensor_dim0[15:0]
  g1[2] = (int)(((tensor_d0 >> 16) & 0xffffu) | ((tensor_d1 & 0xffffu) << 16));
  g1[3] = (int)(((tensor_d1 >> 16) & 0xffffu) | ((tile_d0 & 0xffffu) << 16));
  g1[4] = (int)(tile_d1 & 0xffffu);                             // tile_dim2 = 0
  g1[5] = (int)(d0_stride_units & 0xffffffffull);
  g1[6] = (int)((d0_stride_units >> 32) & 0xffffull);           // dim1_stride = 0 (2D)
  g1[7] = 0;
  i32x4 z4 = {0, 0, 0, 0};
#if defined(__clang_major__) && (__clang_major__ >= 23)
  i32x8 z8 = {0, 0, 0, 0, 0, 0, 0, 0};
  __builtin_amdgcn_tensor_load_to_lds(g0, g1, z4, z4, z8, 0);
#else
  __builtin_amdgcn_tensor_load_to_lds(g0, g1, z4, z4, 0);
#endif
}

// ---------------------------------------------------------------------------
// fp32 -> fp16 conversion
// ---------------------------------------------------------------------------
__global__ void cvt_f32_f16(const float* __restrict__ in, _Float16* __restrict__ out, int n) {
  int i = (blockIdx.x * 256 + threadIdx.x) * 4;
  if (i < n) {
    float4 f = *(const float4*)(in + i);
    v4h h;
    h[0] = (_Float16)f.x; h[1] = (_Float16)f.y;
    h[2] = (_Float16)f.z; h[3] = (_Float16)f.w;
    *(v4h*)(out + i) = h;
  }
}

// ---------------------------------------------------------------------------
// QKV GEMM: one wave -> 32 rows x 64 cols (2 M-tiles x 4 N-tiles, B reused 2x).
// Scatter into Q [b,h,l,hd] (pre-scaled by 1/sqrt(hd)), K [b,h,l,hd], V^T [b,h,hd,l].
// ---------------------------------------------------------------------------
__global__ void __launch_bounds__(128) qkv_gemm_kernel(
    const _Float16* __restrict__ xh, const _Float16* __restrict__ wh,
    const float* __restrict__ bias,
    _Float16* __restrict__ qh, _Float16* __restrict__ kh, _Float16* __restrict__ vT) {
  const int lane = threadIdx.x & 31;
  const int wave = threadIdx.x >> 5;
  const int task = blockIdx.x * 4 + wave;       // (BL/32)*(EQKV/64) tasks
  const int NG = EQKV / 64;                     // 48 column groups
  const int mt = task / NG;
  const int ng = task % NG;
  const int m0 = mt * 32;
  const int n0 = ng * 64;
  const int hf = lane >> 4;
  const int nl = lane & 15;

  v8f acc[2][4] = {};
  const _Float16* arow0 = xh + (size_t)(m0 + nl) * DIMX;
  const _Float16* arow1 = xh + (size_t)(m0 + 16 + nl) * DIMX;
  for (int kk = 0; kk < DIMX; kk += 32) {
    v16h af0 = load_a_frag(arow0 + kk, hf);
    v16h af1 = load_a_frag(arow1 + kk, hf);
#pragma unroll
    for (int t = 0; t < 4; ++t) {
      v16h bf = load_b_frag(wh + (size_t)(n0 + 16 * t + nl) * DIMX + kk, hf);
      acc[0][t] = wmma32(af0, bf, acc[0][t]);
      acc[1][t] = wmma32(af1, bf, acc[1][t]);
    }
  }

  const int s = n0 >> 10;             // 0=Q, 1=K, 2=V
  const int h = (n0 & 1023) >> 6;     // head
#pragma unroll
  for (int t = 0; t < 4; ++t) {
    const float bv = bias[n0 + 16 * t + nl];
    const int d = 16 * t + nl;
#pragma unroll
    for (int mtile = 0; mtile < 2; ++mtile) {
#pragma unroll
      for (int r = 0; r < 8; ++r) {
        const int row = m0 + 16 * mtile + r + 8 * hf;   // global (b*L + l)
        const int b = row >> 11;
        const int l = row & (LSEQ - 1);
        const float v = acc[mtile][t][r] + bv;
        const size_t bh = (size_t)(b * NHEADS + h);
        if (s == 0)      qh[(bh * LSEQ + l) * HDIM + d] = (_Float16)(v * 0.125f);
        else if (s == 1) kh[(bh * LSEQ + l) * HDIM + d] = (_Float16)v;
        else             vT[(bh * HDIM + d) * LSEQ + l] = (_Float16)v;
      }
    }
  }
}

// ---------------------------------------------------------------------------
// Flash attention: 1 block = 4 waves = 64 query rows of one (b,h).
// K (32x64) and V^T (64x32) tiles are DMA'd into LDS by the Tensor Data Mover,
// double-buffered (TDM of block j+1 overlaps compute of block j).
// Q fragments live in VGPRs for the whole stream.
// ---------------------------------------------------------------------------
__global__ void __launch_bounds__(128) attn_kernel(
    const _Float16* __restrict__ qh, const _Float16* __restrict__ kh,
    const _Float16* __restrict__ vT, _Float16* __restrict__ yh) {
  __shared__ __align__(16) _Float16 ldsK[2][32][64];   // [buf][key][hd]
  __shared__ __align__(16) _Float16 ldsV[2][64][32];   // [buf][hd][key]
  __shared__ __align__(16) _Float16 ldsP[4][16][32];   // per-wave P transpose

  const int lane = threadIdx.x & 31;
  const int wave = threadIdx.x >> 5;
  const int hf = lane >> 4;
  const int nl = lane & 15;
  const int qt = blockIdx.x & 31;               // L/64 = 32 q tiles
  const int bh = blockIdx.x >> 5;               // b*NHEADS + h
  const int q0 = qt * 64 + wave * 16;

  const _Float16* qbase = qh + ((size_t)bh * LSEQ + q0 + nl) * HDIM;
  const _Float16* kbase = kh + (size_t)bh * LSEQ * HDIM;
  const _Float16* vtb   = vT + (size_t)bh * HDIM * LSEQ;

  // Q fragments: loaded once, reused every key block.
  v16h qa0 = load_a_frag(qbase + 0, hf);
  v16h qa1 = load_a_frag(qbase + 32, hf);

  v8f o[4] = {};
  float mi[8], li[8];
#pragma unroll
  for (int r = 0; r < 8; ++r) { mi[r] = -1e30f; li[r] = 0.0f; }

  const int nblk = 2 * qt + 2;                  // block-uniform causal bound

  if (wave == 0) {                              // prologue DMA for block 0
    tdm_load_tile(&ldsK[0][0][0], kbase, 512, 1, 512, 1, 512);            // 4KB contiguous
    tdm_load_tile(&ldsV[0][0][0], vtb, 8, 64, 8, 64, (LSEQ * 2) / 8);     // 64 rows x 64B
  }

  for (int blk = 0; blk < nblk; ++blk) {
    const int kb = blk * 32;
    const int cur = blk & 1;
    if (wave == 0) {
      if (blk + 1 < nblk) {                     // prefetch next block's K/V tiles
        tdm_load_tile(&ldsK[cur ^ 1][0][0], kbase + (size_t)(kb + 32) * HDIM,
                      512, 1, 512, 1, 512);
        tdm_load_tile(&ldsV[cur ^ 1][0][0], vtb + (kb + 32),
                      8, 64, 8, 64, (LSEQ * 2) / 8);
        __builtin_amdgcn_s_wait_tensorcnt(2);   // current pair done, next in flight
      } else {
        __builtin_amdgcn_s_wait_tensorcnt(0);
      }
    }
    __syncthreads();                            // publish current K/V tiles

    v8f s0 = {}, s1 = {};                       // S = Q K^T from LDS tiles
#pragma unroll
    for (int kkI = 0; kkI < 2; ++kkI) {
      v16h qa = kkI ? qa1 : qa0;
      v16h b0 = load_b_frag(&ldsK[cur][nl][kkI * 32], hf);
      v16h b1 = load_b_frag(&ldsK[cur][16 + nl][kkI * 32], hf);
      s0 = wmma32(qa, b0, s0);
      s1 = wmma32(qa, b1, s1);
    }

#pragma unroll
    for (int r = 0; r < 8; ++r) {               // causal mask + online softmax
      const int qrow = q0 + r + 8 * hf;
      float a0 = (kb + nl      <= qrow) ? s0[r] : -1e30f;
      float a1 = (kb + 16 + nl <= qrow) ? s1[r] : -1e30f;
      float mx = fmaxf(a0, a1);
#pragma unroll
      for (int off = 1; off < 16; off <<= 1)
        mx = fmaxf(mx, __shfl_xor(mx, off, 32));
      const float mnew = fmaxf(mi[r], mx);
      const float alpha = __expf(mi[r] - mnew);
      const float p0 = __expf(a0 - mnew);
      const float p1 = __expf(a1 - mnew);
      float rs = p0 + p1;
#pragma unroll
      for (int off = 1; off < 16; off <<= 1)
        rs += __shfl_xor(rs, off, 32);
      li[r] = li[r] * alpha + rs;
      mi[r] = mnew;
#pragma unroll
      for (int t = 0; t < 4; ++t) o[t][r] *= alpha;
      ldsP[wave][r + 8 * hf][nl]      = (_Float16)p0;   // C-layout -> row-major tile
      ldsP[wave][r + 8 * hf][16 + nl] = (_Float16)p1;
    }
    asm volatile("s_wait_dscnt 0x0" ::: "memory");      // wave-private LDS handoff
    v16h pa = load_a_frag(&ldsP[wave][nl][0], hf);      // reload P in A-layout
#pragma unroll
    for (int t = 0; t < 4; ++t) {                       // O += P V
      v16h vf = load_b_frag(&ldsV[cur][16 * t + nl][0], hf);
      o[t] = wmma32(pa, vf, o[t]);
    }
    __syncthreads();                            // all waves done with buffer `cur`
  }

  const int b = bh >> 4, h = bh & 15;
#pragma unroll
  for (int t = 0; t < 4; ++t) {
#pragma unroll
    for (int r = 0; r < 8; ++r) {
      const int row = q0 + r + 8 * hf;
      const float v = o[t][r] / li[r];
      yh[((size_t)b * LSEQ + row) * DIMX + h * HDIM + 16 * t + nl] = (_Float16)v;
    }
  }
}

// ---------------------------------------------------------------------------
// Output projection: out = y @ proj_w^T + proj_b (fp32), 32x64 per wave.
// ---------------------------------------------------------------------------
__global__ void __launch_bounds__(128) proj_gemm_kernel(
    const _Float16* __restrict__ yh, const _Float16* __restrict__ wh,
    const float* __restrict__ bias, float* __restrict__ out) {
  const int lane = threadIdx.x & 31;
  const int wave = threadIdx.x >> 5;
  const int task = blockIdx.x * 4 + wave;       // (BL/32)*(DIMX/64)
  const int NG = DIMX / 64;                     // 16
  const int mt = task / NG;
  const int ng = task % NG;
  const int m0 = mt * 32;
  const int n0 = ng * 64;
  const int hf = lane >> 4;
  const int nl = lane & 15;

  v8f acc[2][4] = {};
  const _Float16* arow0 = yh + (size_t)(m0 + nl) * DIMX;
  const _Float16* arow1 = yh + (size_t)(m0 + 16 + nl) * DIMX;
  for (int kk = 0; kk < DIMX; kk += 32) {
    v16h af0 = load_a_frag(arow0 + kk, hf);
    v16h af1 = load_a_frag(arow1 + kk, hf);
#pragma unroll
    for (int t = 0; t < 4; ++t) {
      v16h bf = load_b_frag(wh + (size_t)(n0 + 16 * t + nl) * DIMX + kk, hf);
      acc[0][t] = wmma32(af0, bf, acc[0][t]);
      acc[1][t] = wmma32(af1, bf, acc[1][t]);
    }
  }
#pragma unroll
  for (int t = 0; t < 4; ++t) {
    const float bv = bias[n0 + 16 * t + nl];
#pragma unroll
    for (int mtile = 0; mtile < 2; ++mtile) {
#pragma unroll
      for (int r = 0; r < 8; ++r) {
        const int row = m0 + 16 * mtile + r + 8 * hf;
        out[(size_t)row * DIMX + n0 + 16 * t + nl] = acc[mtile][t][r] + bv;
      }
    }
  }
}

// ---------------------------------------------------------------------------
extern "C" void kernel_launch(void* const* d_in, const int* in_sizes, int n_in,
                              void* d_out, int out_size, void* d_ws, size_t ws_size,
                              hipStream_t stream) {
  (void)in_sizes; (void)n_in; (void)out_size; (void)ws_size;
  const float* x      = (const float*)d_in[0];
  // d_in[1] = attn_mask (causal tril) -- implied analytically, not read
  const float* qkv_w  = (const float*)d_in[2];
  const float* qkv_b  = (const float*)d_in[3];
  const float* proj_w = (const float*)d_in[4];
  const float* proj_b = (const float*)d_in[5];
  float* out = (float*)d_out;

  _Float16* ws = (_Float16*)d_ws;
  size_t off = 0;
  _Float16* xh    = ws + off; off += (size_t)BL * DIMX;     // x in f16
  _Float16* wqkv  = ws + off; off += (size_t)EQKV * DIMX;   // qkv_w f16
  _Float16* wproj = ws + off; off += (size_t)DIMX * DIMX;   // proj_w f16
  _Float16* qh    = ws + off; off += (size_t)BL * DIMX;     // Q [b,h,l,hd] (pre-scaled)
  _Float16* kh    = ws + off; off += (size_t)BL * DIMX;     // K [b,h,l,hd]
  _Float16* vT    = ws + off; off += (size_t)BL * DIMX;     // V^T [b,h,hd,l]
  _Float16* yh    = ws + off; off += (size_t)BL * DIMX;     // attn out [b,l,d] f16

  cvt_f32_f16<<<(BL * DIMX / 4 + 255) / 256, 256, 0, stream>>>(x, xh, BL * DIMX);
  cvt_f32_f16<<<(EQKV * DIMX / 4 + 255) / 256, 256, 0, stream>>>(qkv_w, wqkv, EQKV * DIMX);
  cvt_f32_f16<<<(DIMX * DIMX / 4 + 255) / 256, 256, 0, stream>>>(proj_w, wproj, DIMX * DIMX);

  qkv_gemm_kernel<<<(BL / 32) * (EQKV / 64) / 4, 128, 0, stream>>>(xh, wqkv, qkv_b, qh, kh, vT);
  attn_kernel<<<BBATCH * NHEADS * (LSEQ / 64), 128, 0, stream>>>(qh, kh, vT, yh);
  proj_gemm_kernel<<<(BL / 32) * (DIMX / 64) / 4, 128, 0, stream>>>(yh, wproj, proj_b, out);
}
